// Attention_84567906058480
// MI455X (gfx1250) — matrix operand order repaired and checked by reading, not compiled, not run on hardware
//
#include <hip/hip_runtime.h>
#include <hip/hip_bf16.h>

typedef __bf16 bf16_t;
typedef __attribute__((ext_vector_type(8)))  __bf16 v8bf;
typedef __attribute__((ext_vector_type(16))) __bf16 v16bf;
typedef __attribute__((ext_vector_type(8)))  float  v8f;
typedef __attribute__((ext_vector_type(4)))  unsigned u32x4;
typedef __attribute__((ext_vector_type(8)))  int      i32x8;
typedef __attribute__((ext_vector_type(4)))  int      i32x4;

#define BATCH 4
#define TSEQ  2048
#define CDIM  1024
#define NHEAD 16
#define HDIM  64
#define MROWS (BATCH * TSEQ)   // 8192
#define ATT_SCALE 0.125f       // 64^-0.5

#if __has_builtin(__builtin_amdgcn_tensor_load_to_lds) && __has_builtin(__builtin_amdgcn_s_wait_tensorcnt)
#define USE_TDM 1
#endif

__device__ __forceinline__ bf16_t f2bf(float f) {
    unsigned u = __builtin_bit_cast(unsigned, f);
    unsigned r = (u + 0x7FFFu + ((u >> 16) & 1u)) >> 16;
    return __builtin_bit_cast(bf16_t, (unsigned short)r);
}

__device__ __forceinline__ v16bf cat16(v8bf a, v8bf b) {
    return __builtin_shufflevector(a, b, 0,1,2,3,4,5,6,7,8,9,10,11,12,13,14,15);
}

#ifdef USE_TDM
// 2D tensor tile -> LDS via Tensor Data Mover.
// Tile: tile_d0 x tile_d1 elements of 2 bytes; LDS rows padded:
// pad_interval code 3 (every 16 DWORDs = 64B = 32 bf16), pad_amount code 3 (4 DWORDs = 8 bf16)
// -> LDS row pitch 40 bf16 (80B), matching the compute-side layout.
__device__ __forceinline__ void tdm_load_tile_2d(unsigned lds_off, unsigned long long gaddr,
                                                 unsigned tensor_d0, unsigned tensor_d1,
                                                 unsigned tile_d0, unsigned tile_d1,
                                                 unsigned stride0_elems) {
    u32x4 g0;
    g0[0] = 1u;                                            // count=1, user descriptor
    g0[1] = lds_off;                                       // lds_addr (bytes)
    g0[2] = (unsigned)(gaddr & 0xFFFFFFFFull);             // global_addr[31:0]
    g0[3] = (unsigned)((gaddr >> 32) & 0x1FFFFFFull) | (2u << 30);  // addr[56:32] | type=2
    i32x8 g1;
    g1[0] = (int)((1u << 16) | (1u << 20) | (3u << 22) | (3u << 25)); // 2B elems, pad 16dw/4dw
    g1[1] = (int)((tensor_d0 & 0xFFFFu) << 16);
    g1[2] = (int)(((tensor_d0 >> 16) & 0xFFFFu) | ((tensor_d1 & 0xFFFFu) << 16));
    g1[3] = (int)(((tensor_d1 >> 16) & 0xFFFFu) | ((tile_d0 & 0xFFFFu) << 16));
    g1[4] = (int)(tile_d1 & 0xFFFFu);                      // tile_dim1 (tile_dim2=0)
    g1[5] = (int)stride0_elems;                            // tensor_dim0_stride[31:0]
    g1[6] = 0;
    g1[7] = 0;
    i32x4 z4 = {0, 0, 0, 0};
#if __clang_major__ >= 23
    i32x8 z8 = {0, 0, 0, 0, 0, 0, 0, 0};
    __builtin_amdgcn_tensor_load_to_lds(g0, g1, z4, z4, z8, 0);
#else
    __builtin_amdgcn_tensor_load_to_lds(g0, g1, z4, z4, 0);
#endif
}
#endif

// ---------------- fp32 -> bf16 convert (8 elems / thread) ----------------
__global__ void cvt_bf16(const float4* __restrict__ in, v8bf* __restrict__ out, int n8) {
    int i = blockIdx.x * blockDim.x + threadIdx.x;
    if (i < n8) {
        float4 a = in[2 * i], b = in[2 * i + 1];
        v8bf r;
        r[0] = f2bf(a.x); r[1] = f2bf(a.y); r[2] = f2bf(a.z); r[3] = f2bf(a.w);
        r[4] = f2bf(b.x); r[5] = f2bf(b.y); r[6] = f2bf(b.z); r[7] = f2bf(b.w);
        out[i] = r;
    }
}

// ---------------- GEMM: Y = A[M,K] @ W[N,K]^T ----------------
// MODE 0: bf16 out [B,H,T,D] (scaleA applied)   (Q / K)
// MODE 1: bf16 out [B,H,D,T]                    (V transposed)
// MODE 2: fp32 out [M,N] + bias                 (final projection)
template <int MODE>
__global__ __launch_bounds__(256)
void gemm_bf16(const bf16_t* __restrict__ A, const bf16_t* __restrict__ W,
               bf16_t* __restrict__ outb, float* __restrict__ outf,
               const float* __restrict__ bias, float scaleA, int nrowsW) {
    const int K  = CDIM;
    const int KC = K / 32;
    __shared__ bf16_t As[2][128 * 40];
    __shared__ bf16_t Bs[2][128 * 40];

    const int tid  = threadIdx.x;
    const int MB   = blockIdx.y * 128;
    const int NB   = blockIdx.x * 128;
    const int wave = tid >> 5, lane = tid & 31;
    const int lm   = lane & 15, lh = lane >> 4;
    const int m0   = (wave & 3) * 32;      // 4 row groups of 32
    const int n0   = (wave >> 2) * 64;     // 2 col groups of 64

    v8f acc[2][4] = {};

#ifdef USE_TDM
    const unsigned long long gA = (unsigned long long)(uintptr_t)(const void*)(A + (size_t)MB * K);
    const unsigned long long gW = (unsigned long long)(uintptr_t)(const void*)(W + (size_t)NB * K);
    if (wave == 0) {
        tdm_load_tile_2d((unsigned)(uintptr_t)(void*)&As[0][0], gA, K, MROWS, 32, 128, K);
        tdm_load_tile_2d((unsigned)(uintptr_t)(void*)&Bs[0][0], gW, K, nrowsW, 32, 128, K);
        __builtin_amdgcn_s_wait_tensorcnt(0);
    }
    __syncthreads();
#else
    const int lrow = tid >> 1, lseg = tid & 1;   // 128 rows x 2 segments of 16 bf16
    {
        const v8bf* ga = (const v8bf*)(A + (size_t)(MB + lrow) * K + lseg * 16);
        v8bf a0 = ga[0], a1 = ga[1];
        const v8bf* gb = (const v8bf*)(W + (size_t)(NB + lrow) * K + lseg * 16);
        v8bf b0 = gb[0], b1 = gb[1];
        *(v8bf*)&As[0][lrow * 40 + lseg * 16]     = a0;
        *(v8bf*)&As[0][lrow * 40 + lseg * 16 + 8] = a1;
        *(v8bf*)&Bs[0][lrow * 40 + lseg * 16]     = b0;
        *(v8bf*)&Bs[0][lrow * 40 + lseg * 16 + 8] = b1;
        __syncthreads();
    }
#endif

    for (int kc = 0; kc < KC; ++kc) {
        const int cur = kc & 1;
        const bool more = (kc + 1) < KC;

#ifdef USE_TDM
        if (wave == 0 && more) {
            tdm_load_tile_2d((unsigned)(uintptr_t)(void*)&As[1 - cur][0],
                             gA + (size_t)(kc + 1) * 64, K, MROWS, 32, 128, K);
            tdm_load_tile_2d((unsigned)(uintptr_t)(void*)&Bs[1 - cur][0],
                             gW + (size_t)(kc + 1) * 64, K, nrowsW, 32, 128, K);
        }
#else
        v8bf a0, a1, b0, b1;
        if (more) {
            const v8bf* ga = (const v8bf*)(A + (size_t)(MB + lrow) * K + (kc + 1) * 32 + lseg * 16);
            a0 = ga[0]; a1 = ga[1];
            const v8bf* gb = (const v8bf*)(W + (size_t)(NB + lrow) * K + (kc + 1) * 32 + lseg * 16);
            b0 = gb[0]; b1 = gb[1];
        }
#endif

        // compute on current buffer
        v16bf af[2], bf[4];
        const int kb  = 8 * lh;     // A-frag: k chunks kb..kb+7 and 16+kb..16+kb+7
        const int kb2 = 8 * lh;     // B-frag: k chunks kb2..kb2+7 and 16+kb2..16+kb2+7
#pragma unroll
        for (int mt = 0; mt < 2; ++mt) {
            const bf16_t* p = &As[cur][(m0 + mt * 16 + lm) * 40];
            af[mt] = cat16(*(const v8bf*)(p + kb), *(const v8bf*)(p + 16 + kb));
        }
#pragma unroll
        for (int nt = 0; nt < 4; ++nt) {
            const bf16_t* p = &Bs[cur][(n0 + nt * 16 + lm) * 40 + kb2];
            bf[nt] = cat16(*(const v8bf*)(p), *(const v8bf*)(p + 16));
        }
#pragma unroll
        for (int mt = 0; mt < 2; ++mt)
#pragma unroll
            for (int nt = 0; nt < 4; ++nt)
                acc[mt][nt] = __builtin_amdgcn_wmma_f32_16x16x32_bf16(
                    false, af[mt], false, bf[nt], (short)0, acc[mt][nt], false, false);

#ifdef USE_TDM
        if (wave == 0 && more) __builtin_amdgcn_s_wait_tensorcnt(0);
#else
        if (more) {
            *(v8bf*)&As[1 - cur][lrow * 40 + lseg * 16]     = a0;
            *(v8bf*)&As[1 - cur][lrow * 40 + lseg * 16 + 8] = a1;
            *(v8bf*)&Bs[1 - cur][lrow * 40 + lseg * 16]     = b0;
            *(v8bf*)&Bs[1 - cur][lrow * 40 + lseg * 16 + 8] = b1;
        }
#endif
        __syncthreads();
    }

    // epilogue: C tile layout lane->N, vgpr->M (lanes 16-31: M+8)
#pragma unroll
    for (int mt = 0; mt < 2; ++mt)
#pragma unroll
        for (int nt = 0; nt < 4; ++nt)
#pragma unroll
            for (int r = 0; r < 8; ++r) {
                float v = acc[mt][nt][r];
                int gm = MB + m0 + mt * 16 + r + 8 * lh;
                int gn = NB + n0 + nt * 16 + lm;
                if (MODE == 0) {
                    int bq = gm >> 11, t = gm & (TSEQ - 1);
                    int h = gn >> 6, d = gn & (HDIM - 1);
                    outb[(((size_t)(bq * NHEAD + h)) * TSEQ + t) * HDIM + d] = f2bf(v * scaleA);
                } else if (MODE == 1) {
                    int bq = gm >> 11, t = gm & (TSEQ - 1);
                    int h = gn >> 6, d = gn & (HDIM - 1);
                    outb[(((size_t)(bq * NHEAD + h)) * HDIM + d) * TSEQ + t] = f2bf(v);
                } else {
                    outf[(size_t)gm * CDIM + gn] = v + bias[gn];
                }
            }
}

// ---------------- flash attention over one (b,h), 64 query rows / WG ----------------
__global__ __launch_bounds__(128)
void attn_kernel(const bf16_t* __restrict__ Q, const bf16_t* __restrict__ Kk,
                 const bf16_t* __restrict__ Vt, bf16_t* __restrict__ AO) {
    __shared__ bf16_t Pbuf[4][16 * 40];

    const int qi   = blockIdx.x & (TSEQ / 64 - 1);   // query tile
    const int bh   = blockIdx.x >> 5;                // (b*H + h)
    const size_t base = (size_t)bh * TSEQ * HDIM;

    const int tid = threadIdx.x, wave = tid >> 5, lane = tid & 31;
    const int lm = lane & 15, lh = lane >> 4;
    const int kb = 8 * lh;

    // resident Q fragments (SCALE pre-folded), A-frag layout, k = d
    const bf16_t* qp = Q + base + (size_t)(qi * 64 + wave * 16 + lm) * HDIM;
    v16bf qf0 = cat16(*(const v8bf*)(qp + kb),      *(const v8bf*)(qp + kb + 16));
    v16bf qf1 = cat16(*(const v8bf*)(qp + 32 + kb), *(const v8bf*)(qp + 32 + kb + 16));

    float rm[8], rl[8];
#pragma unroll
    for (int r = 0; r < 8; ++r) { rm[r] = -__builtin_inff(); rl[r] = 0.0f; }
    v8f o[4] = {};
    const v8f vzero = {};

    for (int j = 0; j < TSEQ; j += 32) {
        if (j + 32 < TSEQ) {  // prefetch next K/V tiles -> global_prefetch_b8
            __builtin_prefetch((const void*)(Kk + base + (size_t)(j + 32 + lane) * HDIM), 0, 0);
            __builtin_prefetch((const void*)(Vt + base + (size_t)(lane * 2) * TSEQ + j + 32), 0, 0);
        }

        // K^T B-fragments: n = key, k = d contiguous in K[B,H,T,D]
        v8f sc[2];
#pragma unroll
        for (int nt = 0; nt < 2; ++nt) {
            const bf16_t* kp = Kk + base + (size_t)(j + nt * 16 + lm) * HDIM + 8 * lh;
            v16bf k0 = cat16(*(const v8bf*)(kp),      *(const v8bf*)(kp + 16));
            v16bf k1 = cat16(*(const v8bf*)(kp + 32), *(const v8bf*)(kp + 48));
            v8f t0 = __builtin_amdgcn_wmma_f32_16x16x32_bf16(false, qf0, false, k0,
                                                             (short)0, vzero, false, false);
            sc[nt] = __builtin_amdgcn_wmma_f32_16x16x32_bf16(false, qf1, false, k1,
                                                             (short)0, t0, false, false);
        }

        // online softmax (row r lives in vgpr r, lane half lh; cols across 16 lanes)
        float nm[8], alpha[8];
#pragma unroll
        for (int r = 0; r < 8; ++r) {
            float mx = fmaxf(sc[0][r], sc[1][r]);
            mx = fmaxf(mx, __shfl_xor(mx, 8, 16));
            mx = fmaxf(mx, __shfl_xor(mx, 4, 16));
            mx = fmaxf(mx, __shfl_xor(mx, 2, 16));
            mx = fmaxf(mx, __shfl_xor(mx, 1, 16));
            nm[r]    = fmaxf(rm[r], mx);
            alpha[r] = __expf(rm[r] - nm[r]);
            rm[r]    = nm[r];
        }
#pragma unroll
        for (int r = 0; r < 8; ++r) {
            sc[0][r] = __expf(sc[0][r] - nm[r]);
            sc[1][r] = __expf(sc[1][r] - nm[r]);
            float s = sc[0][r] + sc[1][r];
            s += __shfl_xor(s, 8, 16);
            s += __shfl_xor(s, 4, 16);
            s += __shfl_xor(s, 2, 16);
            s += __shfl_xor(s, 1, 16);
            rl[r] = rl[r] * alpha[r] + s;
        }
#pragma unroll
        for (int nt = 0; nt < 4; ++nt)
#pragma unroll
            for (int r = 0; r < 8; ++r) o[nt][r] *= alpha[r];

        // P (C layout) -> LDS -> A-fragment layout (per-wave buffer, in-wave DS ordering)
        bf16_t* pb = Pbuf[wave];
#pragma unroll
        for (int t = 0; t < 2; ++t)
#pragma unroll
            for (int r = 0; r < 8; ++r)
                pb[(r + 8 * lh) * 40 + t * 16 + lm] = f2bf(sc[t][r]);
        const bf16_t* pr = pb + lm * 40;
        v16bf pf = cat16(*(const v8bf*)(pr + kb), *(const v8bf*)(pr + 16 + kb));

        // O += P @ V : B-frag n = d, k = key contiguous in Vt[B,H,D,T]
#pragma unroll
        for (int nt = 0; nt < 4; ++nt) {
            const bf16_t* vp = Vt + base + (size_t)(nt * 16 + lm) * TSEQ + j + 8 * lh;
            v16bf vf = cat16(*(const v8bf*)(vp), *(const v8bf*)(vp + 16));
            o[nt] = __builtin_amdgcn_wmma_f32_16x16x32_bf16(false, pf, false, vf,
                                                            (short)0, o[nt], false, false);
        }
    }

    // normalize and write [B,T,C] bf16
    const int b = bh >> 4, h = bh & (NHEAD - 1);
    float rinv[8];
#pragma unroll
    for (int r = 0; r < 8; ++r) rinv[r] = 1.0f / rl[r];
#pragma unroll
    for (int nt = 0; nt < 4; ++nt)
#pragma unroll
        for (int r = 0; r < 8; ++r) {
            int t = qi * 64 + wave * 16 + r + 8 * lh;
            int c = h * HDIM + nt * 16 + lm;
            AO[((size_t)b * TSEQ + t) * CDIM + c] = f2bf(o[nt][r] * rinv[r]);
        }
}

extern "C" void kernel_launch(void* const* d_in, const int* in_sizes, int n_in,
                              void* d_out, int out_size, void* d_ws, size_t ws_size,
                              hipStream_t stream) {
    const float* hs = (const float*)d_in[0];
    const float* wq = (const float*)d_in[1];
    const float* wk = (const float*)d_in[2];
    const float* wv = (const float*)d_in[3];
    const float* wo = (const float*)d_in[4];
    const float* bo = (const float*)d_in[5];
    float* out = (float*)d_out;

    char* ws = (char*)d_ws;
    const size_t MB1 = 1ull << 20;
    bf16_t* hb  = (bf16_t*)(ws);                 // 16 MB  [B*T, C]
    bf16_t* wqb = (bf16_t*)(ws + 16 * MB1);      //  2 MB
    bf16_t* wkb = (bf16_t*)(ws + 18 * MB1);
    bf16_t* wvb = (bf16_t*)(ws + 20 * MB1);
    bf16_t* wob = (bf16_t*)(ws + 22 * MB1);
    bf16_t* Qb  = (bf16_t*)(ws + 24 * MB1);      // 16 MB  [B,H,T,D] (pre-scaled)
    bf16_t* Kb  = (bf16_t*)(ws + 40 * MB1);      // 16 MB  [B,H,T,D]
    bf16_t* Vtb = (bf16_t*)(ws + 56 * MB1);      // 16 MB  [B,H,D,T]
    bf16_t* AOb = (bf16_t*)(ws + 72 * MB1);      // 16 MB  [B,T,C]

    const int nh8 = MROWS * CDIM / 8;    // 1M vec8
    const int nw8 = CDIM * CDIM / 8;     // 128K vec8
    cvt_bf16<<<(nh8 + 255) / 256, 256, 0, stream>>>((const float4*)hs, (v8bf*)hb, nh8);
    cvt_bf16<<<(nw8 + 255) / 256, 256, 0, stream>>>((const float4*)wq, (v8bf*)wqb, nw8);
    cvt_bf16<<<(nw8 + 255) / 256, 256, 0, stream>>>((const float4*)wk, (v8bf*)wkb, nw8);
    cvt_bf16<<<(nw8 + 255) / 256, 256, 0, stream>>>((const float4*)wv, (v8bf*)wvb, nw8);
    cvt_bf16<<<(nw8 + 255) / 256, 256, 0, stream>>>((const float4*)wo, (v8bf*)wob, nw8);

    dim3 ggrid(CDIM / 128, MROWS / 128);         // 8 x 64
    gemm_bf16<0><<<ggrid, 256, 0, stream>>>(hb, wqb, Qb,  nullptr, nullptr, ATT_SCALE, CDIM);
    gemm_bf16<0><<<ggrid, 256, 0, stream>>>(hb, wkb, Kb,  nullptr, nullptr, 1.0f, CDIM);
    gemm_bf16<1><<<ggrid, 256, 0, stream>>>(hb, wvb, Vtb, nullptr, nullptr, 1.0f, CDIM);

    attn_kernel<<<BATCH * NHEAD * (TSEQ / 64), 128, 0, stream>>>(Qb, Kb, Vtb, AOb);

    gemm_bf16<2><<<ggrid, 256, 0, stream>>>(AOb, wob, nullptr, out, bo, 1.0f, CDIM);
}
